// D_MoE_Block_38946763440634
// MI455X (gfx1250) — matrix operand order repaired and checked by reading, not compiled, not run on hardware
//
#include <hip/hip_runtime.h>

// ---------------- problem constants ----------------
constexpr int B_  = 4;
constexpr int C_  = 96;     // channels
constexpr int H_  = 256;
constexpr int W_  = 256;
constexpr int HW_ = H_ * W_;
constexpr int C2_ = 192;    // ffn hidden

typedef __attribute__((ext_vector_type(16))) __bf16 v16bf;
typedef __attribute__((ext_vector_type(8)))  __bf16 v8bf;
typedef __attribute__((ext_vector_type(4)))  __bf16 v4bf;
typedef __attribute__((ext_vector_type(8)))  float  v8f;
typedef __attribute__((ext_vector_type(4)))  float  f4;

// ---------------- LDS geometry ----------------
// kernel 1: one image row (256 px)
constexpr int PA1 = 104;                                  // [m][k] bf16 pitch (52 dwords: bank-tiling exact)
constexpr int PB  = 104;                                  // [n][k<=96] bf16 pitch
constexpr int SM1 = 256 * PA1 * 2 + C_ * PB * 2;          // lds_a + lds_w = 73,216 B

// kernel 3: 128-px tile
constexpr int P3   = 128;
constexpr int PA3  = 104;                                 // [m][k<=96] bf16
constexpr int PB2  = 200;                                 // [n][k<=192] / [m][k<=192] bf16 (100 dwords)
constexpr int PX   = 132;                                 // [n][m] fp32 pitch
constexpr int OFF_XA  = 0;                                // bf16 [128][PA3]
constexpr int OFF_PW  = OFF_XA  + P3  * PA3 * 2;          // bf16 [96][PA3]
constexpr int OFF_F1  = OFF_PW  + C_  * PA3 * 2;          // bf16 [192][PA3]
constexpr int OFF_F2  = OFF_F1  + C2_ * PA3 * 2;          // bf16 [96][PB2]
constexpr int OFF_X1B = OFF_F2  + C_  * PB2 * 2;          // bf16 [128][PA3]  (ffn1 A source)
constexpr int OFF_X1F = OFF_X1B + P3  * PA3 * 2;          // fp32 [96][PX]    (residual)
constexpr int OFF_H   = OFF_X1F + C_  * PX  * 4;          // bf16 [128][PB2]  (ffn2 A source)
constexpr int SM3     = OFF_H   + P3  * PB2 * 2;          // 253,440 B

// optional CDNA5 async global->LDS path (compile-safe guard)
#if defined(__gfx1250__) && __has_builtin(__builtin_amdgcn_global_load_async_to_lds_b128)
#define HAVE_ASYNC_LDS 1
typedef __attribute__((ext_vector_type(4))) int i4v;
typedef __attribute__((address_space(1))) i4v* gi4p;   // global int4*
typedef __attribute__((address_space(3))) i4v* li4p;   // LDS int4*
#endif

// float -> bf16 (RNE)
__device__ __forceinline__ __bf16 f2bf(float f) {
    unsigned int u = __builtin_bit_cast(unsigned int, f);
    u += 0x7FFFu + ((u >> 16) & 1u);
    return __builtin_bit_cast(__bf16, (unsigned short)(u >> 16));
}

// ---- fragment loaders: contiguous per-lane reads (2x ds_load_b128 each) ----
// A (16x32 MxK) from [m][k] bf16: lane&15=M, lane>=16 -> K+8; runs at +0 and +16
__device__ __forceinline__ v16bf a_frag(const __bf16* base, int pitch,
                                        int mBase, int kBase, int lane) {
    const __bf16* p = base + (mBase + (lane & 15)) * pitch + kBase + ((lane & 16) ? 8 : 0);
    v8bf lo = *(const v8bf*)p;
    v8bf hi = *(const v8bf*)(p + 16);
    return __builtin_shufflevector(lo, hi, 0,1,2,3,4,5,6,7,8,9,10,11,12,13,14,15);
}
// B (32x16 KxN) from [n][k] bf16: lane&15=N, lane>=16 -> K+8; runs at +0 and +16
__device__ __forceinline__ v16bf b_frag(const __bf16* base, int pitch,
                                        int nBase, int kBase, int lane) {
    const __bf16* p = base + (nBase + (lane & 15)) * pitch + kBase + ((lane & 16) ? 8 : 0);
    v8bf lo = *(const v8bf*)p;
    v8bf hi = *(const v8bf*)(p + 16);
    return __builtin_shufflevector(lo, hi, 0,1,2,3,4,5,6,7,8,9,10,11,12,13,14,15);
}

__device__ __forceinline__ v8f wmma_bf16(v16bf a, v16bf b, v8f c) {
    return __builtin_amdgcn_wmma_f32_16x16x32_bf16(false, a, false, b,
                                                   (short)0, c, false, false);
}

// stage a row-major fp32 weight matrix [rows][cols] into bf16 LDS [rows][pitch]
__device__ __forceinline__ void stage_w(__bf16* dst, const float* src,
                                        int rows, int cols, int pitch, int tid) {
    const int chunks = rows * (cols / 4);
    for (int i = tid; i < chunks; i += 256) {
        const int r  = i / (cols / 4);
        const int c4 = (i % (cols / 4)) * 4;
        f4 w = *(const f4*)&src[r * cols + c4];
        v4bf o; o[0] = f2bf(w.x); o[1] = f2bf(w.y); o[2] = f2bf(w.z); o[3] = f2bf(w.w);
        *(v4bf*)&dst[r * pitch + c4] = o;
    }
}

// =====================================================================
// Kernel 1: LayerNorm(channel) + e0 pointwise GEMM (WMMA bf16)
// =====================================================================
__global__ void k_ln_pw0(const float* __restrict__ x,
                         const float* __restrict__ ln_w, const float* __restrict__ ln_b,
                         const float* __restrict__ pw_w, const float* __restrict__ pw_b,
                         float* __restrict__ xn_out, float* __restrict__ t0_out) {
    extern __shared__ unsigned char smem[];
    __bf16* lds_a = (__bf16*)smem;                         // [256][PA1]  activations [m][k]
    __bf16* lds_w = (__bf16*)(smem + 256 * PA1 * 2);       // [96][PB]    weights     [n][k]

    const int tid = threadIdx.x;            // pixel w in the row
    const int b   = blockIdx.x / H_;
    const int h   = blockIdx.x % H_;
    const size_t rowBase = (size_t)b * C_ * HW_ + (size_t)h * W_;

    // pass 1: per-pixel channel stats (coalesced across lanes per channel)
    float sum = 0.f, sq = 0.f;
    for (int c = 0; c < C_; ++c) {
        float v = x[rowBase + (size_t)c * HW_ + tid];
        sum += v; sq += v * v;
    }
    const float mean = sum * (1.f / C_);
    const float var  = sq * (1.f / C_) - mean * mean;
    const float rstd = rsqrtf(var + 1e-6f);

    // pass 2: normalize; fp32 -> global, packed bf16 row -> LDS (b128 stores)
    for (int c0 = 0; c0 < C_; c0 += 8) {
        v8bf pk;
#pragma unroll
        for (int j = 0; j < 8; ++j) {
            const int c = c0 + j;
            float v   = x[rowBase + (size_t)c * HW_ + tid];
            float xnv = (v - mean) * rstd * ln_w[c] + ln_b[c];
            xn_out[rowBase + (size_t)c * HW_ + tid] = xnv;
            pk[j] = f2bf(xnv);
        }
        *(v8bf*)&lds_a[tid * PA1 + c0] = pk;
    }
    stage_w(lds_w, pw_w, C_, C_, PB, tid);
    __syncthreads();

    // GEMM: M=256 (16 tiles) x N=96 (6 tiles), K=96 (3 steps)
    const int lane = tid & 31, wv = tid >> 5;
    for (int t = wv; t < 16 * 6; t += 8) {
        const int mt = t / 6, nt = t % 6;
        v8f acc = {};
#pragma unroll
        for (int kk = 0; kk < 3; ++kk) {
            acc = wmma_bf16(a_frag(lds_a, PA1, mt * 16, kk * 32, lane),
                            b_frag(lds_w, PB,  nt * 16, kk * 32, lane), acc);
        }
        const int   n    = nt * 16 + (lane & 15);
        const float bias = pw_b[n];
        const int   mlo  = mt * 16 + ((lane & 16) ? 8 : 0);
        f4 lo = { acc[0] + bias, acc[1] + bias, acc[2] + bias, acc[3] + bias };
        f4 hi = { acc[4] + bias, acc[5] + bias, acc[6] + bias, acc[7] + bias };
        float* dst = &t0_out[rowBase + (size_t)n * HW_ + mlo];
        *(f4*)dst       = lo;                 // 2x global_store_b128
        *(f4*)(dst + 4) = hi;
    }
}

// =====================================================================
// Kernel 2: depthwise experts + soft mix + prompt gate -> xa (bf16, channel-last)
// =====================================================================
__global__ void k_dw_mix(const float* __restrict__ xn, const float* __restrict__ t0,
                         const float* __restrict__ e0w, const float* __restrict__ e0b,
                         const float* __restrict__ e1w, const float* __restrict__ e1b,
                         const float* __restrict__ e2w, const float* __restrict__ e2b,
                         const float* __restrict__ scale_w, const float* __restrict__ prompt,
                         __bf16* __restrict__ xa) {
    const int w = threadIdx.x;
    const int b = blockIdx.x / H_;
    const int h = blockIdx.x % H_;
    const float s0 = scale_w[b * 3 + 0];
    const float s1 = scale_w[b * 3 + 1];
    const float s2 = scale_w[b * 3 + 2];
    const size_t pixBase = ((size_t)(b * H_ + h) * W_ + w) * C_;   // channel-last

    for (int c0 = 0; c0 < C_; c0 += 8) {
        v8bf pk;
#pragma unroll
        for (int j = 0; j < 8; ++j) {
            const int c = c0 + j;
            const float* xc = xn + (size_t)(b * C_ + c) * HW_;
            const float* tc = t0 + (size_t)(b * C_ + c) * HW_;
            float e0 = e0b[c], e1 = e1b[c], e2 = e2b[c];
#pragma unroll
            for (int dy = -1; dy <= 1; ++dy)
#pragma unroll
                for (int dx = -1; dx <= 1; ++dx) {
                    int hh = h + dy, ww = w + dx;          // expert0: 3x3 d1 on t0
                    if ((unsigned)hh < (unsigned)H_ && (unsigned)ww < (unsigned)W_)
                        e0 += e0w[c * 9 + (dy + 1) * 3 + (dx + 1)] * tc[(size_t)hh * W_ + ww];
                }
#pragma unroll
            for (int dy = -1; dy <= 1; ++dy)
#pragma unroll
                for (int dx = -1; dx <= 1; ++dx) {
                    int hh = h + 2 * dy, ww = w + 2 * dx;  // expert1: 3x3 d2 on xn
                    if ((unsigned)hh < (unsigned)H_ && (unsigned)ww < (unsigned)W_)
                        e1 += e1w[c * 9 + (dy + 1) * 3 + (dx + 1)] * xc[(size_t)hh * W_ + ww];
                }
#pragma unroll
            for (int dy = -2; dy <= 2; ++dy)
#pragma unroll
                for (int dx = -2; dx <= 2; ++dx) {
                    int hh = h + 3 * dy, ww = w + 3 * dx;  // expert2: 5x5 d3 on xn
                    if ((unsigned)hh < (unsigned)H_ && (unsigned)ww < (unsigned)W_)
                        e2 += e2w[c * 25 + (dy + 2) * 5 + (dx + 2)] * xc[(size_t)hh * W_ + ww];
                }
            const float g = 1.f + prompt[b * C_ + c];
            pk[j] = f2bf((s0 * e0 + s1 * e1 + s2 * e2) * g);
        }
        *(v8bf*)&xa[pixBase + c0] = pk;                    // global_store_b128
    }
}

// =====================================================================
// Kernel 3: fused x1 = x + proj(xa); h = gelu(ffn1(x1)); out = x1 + ffn2(h)
// =====================================================================
__global__ void k_fused_mlp(const __bf16* __restrict__ xa,
                            const float* __restrict__ x,
                            const float* __restrict__ proj_w, const float* __restrict__ proj_b,
                            const float* __restrict__ ffn1_w, const float* __restrict__ ffn1_b,
                            const float* __restrict__ ffn2_w, const float* __restrict__ ffn2_b,
                            float* __restrict__ out) {
    extern __shared__ unsigned char smem[];
    __bf16* lds_xa  = (__bf16*)(smem + OFF_XA);   // [128][PA3] A of proj
    __bf16* lds_pw  = (__bf16*)(smem + OFF_PW);   // [96][PA3]  proj  W [n][k]
    __bf16* lds_f1  = (__bf16*)(smem + OFF_F1);   // [192][PA3] ffn1  W [n][k]
    __bf16* lds_f2  = (__bf16*)(smem + OFF_F2);   // [96][PB2]  ffn2  W [n][k]
    __bf16* lds_x1b = (__bf16*)(smem + OFF_X1B);  // [128][PA3] A of ffn1
    float*  lds_x1f = (float*) (smem + OFF_X1F);  // [96][PX]   x1 fp32 (residual)
    __bf16* lds_h   = (__bf16*)(smem + OFF_H);    // [128][PB2] A of ffn2

    const int tid = threadIdx.x;
    const int blk = blockIdx.x;
    const int b   = blk / (H_ * (W_ / P3));
    const int r   = blk % (H_ * (W_ / P3));
    const int h   = r / (W_ / P3);
    const int w0  = (r % (W_ / P3)) * P3;
    const size_t rowBase = (size_t)b * C_ * HW_ + (size_t)h * W_ + w0;

    // ---- stage xa tile: straight b128 copy (async-to-LDS when available) ----
    const __bf16* xaRow = xa + ((size_t)(b * H_ + h) * W_ + w0) * C_;
    for (int i = tid; i < P3 * (C_ / 8); i += 256) {
        const int m = i / (C_ / 8), j = (i % (C_ / 8)) * 8;
#if defined(HAVE_ASYNC_LDS)
        __builtin_amdgcn_global_load_async_to_lds_b128(
            (gi4p)(xaRow + m * C_ + j),
            (li4p)(lds_xa + m * PA3 + j), 0, 0);
#else
        *(v8bf*)&lds_xa[m * PA3 + j] = *(const v8bf*)&xaRow[m * C_ + j];
#endif
    }
    // ---- stage weights (native [out][in] row-major == B layout) ----
    stage_w(lds_pw, proj_w, C_,  C_,  PA3, tid);
    stage_w(lds_f1, ffn1_w, C2_, C_,  PA3, tid);
    stage_w(lds_f2, ffn2_w, C_,  C2_, PB2, tid);
    // ---- pre-stage shortcut x into residual array [n][m] (coalesced) ----
    for (int i = tid; i < C_ * (P3 / 4); i += 256) {
        const int c = i / (P3 / 4), p4 = (i % (P3 / 4)) * 4;
        *(f4*)&lds_x1f[c * PX + p4] = *(const f4*)&x[rowBase + (size_t)c * HW_ + p4];
    }
#if defined(HAVE_ASYNC_LDS)
    asm volatile("s_wait_asynccnt 0" ::: "memory");
#endif
    __syncthreads();

    const int lane = tid & 31, wv = tid >> 5;

    // ---- GEMM 1: x1 = x + proj(xa) + b   (M=128, N=96, K=96) ----
    for (int t = wv; t < 8 * 6; t += 8) {
        const int mt = t / 6, nt = t % 6;
        v8f acc = {};
#pragma unroll
        for (int kk = 0; kk < 3; ++kk)
            acc = wmma_bf16(a_frag(lds_xa, PA3, mt * 16, kk * 32, lane),
                            b_frag(lds_pw, PA3, nt * 16, kk * 32, lane), acc);
        const int   n    = nt * 16 + (lane & 15);
        const float bias = proj_b[n];
        const int   mlo  = mt * 16 + ((lane & 16) ? 8 : 0);
        f4* px = (f4*)&lds_x1f[n * PX + mlo];
        f4 lo = px[0], hi = px[1];
        lo.x += acc[0] + bias; lo.y += acc[1] + bias; lo.z += acc[2] + bias; lo.w += acc[3] + bias;
        hi.x += acc[4] + bias; hi.y += acc[5] + bias; hi.z += acc[6] + bias; hi.w += acc[7] + bias;
        px[0] = lo; px[1] = hi;                                   // fp32 residual [n][m]
#pragma unroll
        for (int v = 0; v < 8; ++v) {                             // bf16 A copy [m][n]
            float val = (v < 4) ? ((const float*)&lo)[v] : ((const float*)&hi)[v - 4];
            lds_x1b[(mlo + v) * PA3 + n] = f2bf(val);
        }
    }
    __syncthreads();

    // ---- GEMM 2: h = gelu(ffn1(x1) + b)   (M=128, N=192, K=96) ----
    for (int t = wv; t < 8 * 12; t += 8) {
        const int mt = t / 12, nt = t % 12;
        v8f acc = {};
#pragma unroll
        for (int kk = 0; kk < 3; ++kk)
            acc = wmma_bf16(a_frag(lds_x1b, PA3, mt * 16, kk * 32, lane),
                            b_frag(lds_f1,  PA3, nt * 16, kk * 32, lane), acc);
        const int   n    = nt * 16 + (lane & 15);
        const float bias = ffn1_b[n];
        const int   mlo  = mt * 16 + ((lane & 16) ? 8 : 0);
#pragma unroll
        for (int v = 0; v < 8; ++v) {
            float val = acc[v] + bias;
            float g   = 0.5f * val * (1.f + erff(val * 0.70710678118654752f)); // exact gelu
            lds_h[(mlo + v) * PB2 + n] = f2bf(g);                 // bf16 A copy [m][n]
        }
    }
    __syncthreads();

    // ---- GEMM 3: out = x1 + ffn2(h) + b   (M=128, N=96, K=192) ----
    for (int t = wv; t < 8 * 6; t += 8) {
        const int mt = t / 6, nt = t % 6;
        v8f acc = {};
#pragma unroll
        for (int kk = 0; kk < 6; ++kk)
            acc = wmma_bf16(a_frag(lds_h,  PB2, mt * 16, kk * 32, lane),
                            b_frag(lds_f2, PB2, nt * 16, kk * 32, lane), acc);
        const int   n    = nt * 16 + (lane & 15);
        const float bias = ffn2_b[n];
        const int   mlo  = mt * 16 + ((lane & 16) ? 8 : 0);
        const f4* px = (const f4*)&lds_x1f[n * PX + mlo];
        f4 lo = px[0], hi = px[1];
        lo.x += acc[0] + bias; lo.y += acc[1] + bias; lo.z += acc[2] + bias; lo.w += acc[3] + bias;
        hi.x += acc[4] + bias; hi.y += acc[5] + bias; hi.z += acc[6] + bias; hi.w += acc[7] + bias;
        float* dst = &out[rowBase + (size_t)n * HW_ + mlo];
        *(f4*)dst       = lo;                                     // 2x global_store_b128
        *(f4*)(dst + 4) = hi;
    }
}

// =====================================================================
extern "C" void kernel_launch(void* const* d_in, const int* in_sizes, int n_in,
                              void* d_out, int out_size, void* d_ws, size_t ws_size,
                              hipStream_t stream) {
    const float* x       = (const float*)d_in[0];
    const float* prompt  = (const float*)d_in[1];
    const float* scale_w = (const float*)d_in[2];
    const float* ln_w    = (const float*)d_in[3];
    const float* ln_b    = (const float*)d_in[4];
    const float* e0_pw_w = (const float*)d_in[5];
    const float* e0_pw_b = (const float*)d_in[6];
    const float* e0_dw_w = (const float*)d_in[7];
    const float* e0_dw_b = (const float*)d_in[8];
    const float* e1_dw_w = (const float*)d_in[9];
    const float* e1_dw_b = (const float*)d_in[10];
    const float* e2_dw_w = (const float*)d_in[11];
    const float* e2_dw_b = (const float*)d_in[12];
    const float* proj_w  = (const float*)d_in[13];
    const float* proj_b  = (const float*)d_in[14];
    const float* ffn1_w  = (const float*)d_in[15];
    const float* ffn1_b  = (const float*)d_in[16];
    const float* ffn2_w  = (const float*)d_in[17];
    const float* ffn2_b  = (const float*)d_in[18];
    float* out = (float*)d_out;

    // workspace: xn (fp32) | t0 (fp32) | xa (bf16, channel-last)
    const size_t TENS_F32 = (size_t)B_ * C_ * HW_ * sizeof(float);   // 100.7 MB
    char* ws = (char*)d_ws;
    float*  xn = (float*)ws;
    float*  t0 = (float*)(ws + TENS_F32);
    __bf16* xa = (__bf16*)(ws + 2 * TENS_F32);

    (void)hipFuncSetAttribute(reinterpret_cast<const void*>(k_ln_pw0),
                              hipFuncAttributeMaxDynamicSharedMemorySize, SM1);
    (void)hipFuncSetAttribute(reinterpret_cast<const void*>(k_fused_mlp),
                              hipFuncAttributeMaxDynamicSharedMemorySize, SM3);

    k_ln_pw0<<<B_ * H_, 256, SM1, stream>>>(x, ln_w, ln_b, e0_pw_w, e0_pw_b, xn, t0);
    k_dw_mix<<<B_ * H_, 256, 0, stream>>>(xn, t0, e0_dw_w, e0_dw_b, e1_dw_w, e1_dw_b,
                                          e2_dw_w, e2_dw_b, scale_w, prompt, xa);
    k_fused_mlp<<<B_ * H_ * (W_ / P3), 256, SM3, stream>>>(xa, x, proj_w, proj_b,
                                                           ffn1_w, ffn1_b, ffn2_w, ffn2_b, out);
}
